// LinformerAttention_26723286516318
// MI455X (gfx1250) — matrix operand order repaired and checked by reading, not compiled, not run on hardware
//
#include <hip/hip_runtime.h>

#define B_  4
#define H_  12
#define S_  4096
#define D_  64
#define KL_ 256
#define NS_ 64
#define KC_ 320   // KL + NS

typedef __attribute__((ext_vector_type(16))) __bf16          v16bf;
typedef __attribute__((ext_vector_type(8)))  __bf16          v8bf;
typedef __attribute__((ext_vector_type(2)))  __bf16          v2bf;
typedef __attribute__((ext_vector_type(8)))  float           v8f;

struct B32x2 { uint4 lo, hi; };

// Native f32 -> bf16 (fptrunc; lowers to v_cvt_pk_bf16_f32 pairs on gfx1250)
__device__ __forceinline__ unsigned short f2bf(float f) {
  return __builtin_bit_cast(unsigned short, (__bf16)f);
}

__device__ __forceinline__ float f4c(const float4& v, int j) {
  return j == 0 ? v.x : j == 1 ? v.y : j == 2 ? v.z : v.w;
}

__device__ __forceinline__ v8f wmma_bf16(v16bf a, v16bf b, v8f c) {
  // D = A(16x32 bf16) * B(32x16 bf16) + C(16x16 f32)
  return __builtin_amdgcn_wmma_f32_16x16x32_bf16(false, a, false, b,
                                                 (short)0, c, false, false);
}

// A-fragment from fp32 row-major memory. 'p' points at this lane's K-base.
// Elements 0..7 = K k0..k0+7 (p[0..7]); elements 8..15 = K k0+16..k0+23 (p[16..23]).
__device__ __forceinline__ v16bf ld_afrag_f32(const float* p) {
  float4 f0 = *(const float4*)(p + 0);
  float4 f1 = *(const float4*)(p + 4);
  float4 f2 = *(const float4*)(p + 16);
  float4 f3 = *(const float4*)(p + 20);
  v16bf v;
  v[0]  = (__bf16)f0.x; v[1]  = (__bf16)f0.y; v[2]  = (__bf16)f0.z; v[3]  = (__bf16)f0.w;
  v[4]  = (__bf16)f1.x; v[5]  = (__bf16)f1.y; v[6]  = (__bf16)f1.z; v[7]  = (__bf16)f1.w;
  v[8]  = (__bf16)f2.x; v[9]  = (__bf16)f2.y; v[10] = (__bf16)f2.z; v[11] = (__bf16)f2.w;
  v[12] = (__bf16)f3.x; v[13] = (__bf16)f3.y; v[14] = (__bf16)f3.z; v[15] = (__bf16)f3.w;
  return v;
}

// A-fragment from bf16 memory: segments p[0..7] and p[16..23].
__device__ __forceinline__ v16bf ld_afrag_bf16(const unsigned short* p) {
  B32x2 t;
  t.lo = *(const uint4*)(p);
  t.hi = *(const uint4*)(p + 16);
  return __builtin_bit_cast(v16bf, t);
}

// B-fragment from bf16 memory: segments p[0..7] and p[16..23].
__device__ __forceinline__ v16bf ld_bfrag_bf16(const unsigned short* p) {
  B32x2 t;
  t.lo = *(const uint4*)(p);
  t.hi = *(const uint4*)(p + 16);
  return __builtin_bit_cast(v16bf, t);
}

// ---------------------------------------------------------------------------
// Phase 1: Kp = E @ (K*mask), Vp = E @ (V*mask).
// Kc rows 0..255 stored row-major (kc x d) bf16; Vct stored transposed (d x kc).
// Grid: B*H*4 blocks (4 kl-blocks of 64 rows), 256 threads = 8 waves.
// Waves 0-3 do K-projection (16 kl rows each), waves 4-7 do V-projection.
// ---------------------------------------------------------------------------
__global__ __launch_bounds__(256) void proj_kernel(
    const float* __restrict__ K, const float* __restrict__ V,
    const float* __restrict__ mask, const float* __restrict__ E,
    unsigned short* __restrict__ Kc, unsigned short* __restrict__ Vct) {
  __shared__ unsigned short ldsKT[64 * 40];   // transposed bf16 K chunk (d x 32, pad 40)
  __shared__ unsigned short ldsVT[64 * 40];

  const int bh    = blockIdx.x >> 2;
  const int klblk = blockIdx.x & 3;
  const int b = bh / H_;
  const int h = bh - b * H_;
  const int tid  = threadIdx.x;
  const int wave = tid >> 5;
  const int lane = tid & 31;
  const bool isV = wave >= 4;
  const int kl0  = klblk * 64 + (wave & 3) * 16;

  // cooperative staging: thread loads float4 along d for an s-pair
  const int d4 = (tid & 15) * 4;      // 0,4,..,60
  const int sp = (tid >> 4) * 2;      // 0,2,..,30

  const float* __restrict__ Eh = E + (size_t)h * KL_ * S_;
  const int ln    = lane & 15;
  const int arow  = kl0 + ln;
  const int k0    = (lane >> 4) * 8;
  const int hioff = (lane & 16) ? 8 : 0;
  const unsigned short* ldsT = isV ? ldsVT : ldsKT;

  v8f acc[4] = {};

  for (int s0 = 0; s0 < S_; s0 += 32) {
    __syncthreads();
    {
      const int s = s0 + sp;
      const float m0 = mask[b * S_ + s];
      const float m1 = mask[b * S_ + s + 1];
      const float4 kr0 = *(const float4*)(K + ((size_t)bh * S_ + s    ) * D_ + d4);
      const float4 kr1 = *(const float4*)(K + ((size_t)bh * S_ + s + 1) * D_ + d4);
      const float4 vr0 = *(const float4*)(V + ((size_t)bh * S_ + s    ) * D_ + d4);
      const float4 vr1 = *(const float4*)(V + ((size_t)bh * S_ + s + 1) * D_ + d4);
      #pragma unroll
      for (int j = 0; j < 4; ++j) {
        v2bf pk, pv;
        pk[0] = (__bf16)(f4c(kr0, j) * m0);
        pk[1] = (__bf16)(f4c(kr1, j) * m1);
        pv[0] = (__bf16)(f4c(vr0, j) * m0);
        pv[1] = (__bf16)(f4c(vr1, j) * m1);
        *(unsigned*)(&ldsKT[(d4 + j) * 40 + sp]) = __builtin_bit_cast(unsigned, pk);
        *(unsigned*)(&ldsVT[(d4 + j) * 40 + sp]) = __builtin_bit_cast(unsigned, pv);
      }
    }
    __syncthreads();

    if (s0 + 32 < S_)
      __builtin_prefetch(Eh + (size_t)arow * S_ + s0 + 32 + k0, 0, 0);

    const v16bf a = ld_afrag_f32(Eh + (size_t)arow * S_ + s0 + k0);
    // load all 4 B-fragments first (distinct registers -> overlapped DS waits)
    v16bf bfr[4];
    #pragma unroll
    for (int dt = 0; dt < 4; ++dt)
      bfr[dt] = ld_bfrag_bf16(ldsT + (dt * 16 + ln) * 40 + hioff);
    #pragma unroll
    for (int dt = 0; dt < 4; ++dt)
      acc[dt] = wmma_bf16(a, bfr[dt], acc[dt]);
  }

  if (!isV) {
    // Kp -> Kc row-major (kc x d)
    #pragma unroll
    for (int dt = 0; dt < 4; ++dt) {
      #pragma unroll
      for (int r = 0; r < 8; ++r) {
        const int row = kl0 + r + ((lane & 16) ? 8 : 0);
        const int col = dt * 16 + ln;
        Kc[((size_t)bh * KC_ + row) * D_ + col] = f2bf(acc[dt][r]);
      }
    }
  } else {
    // Vp -> Vct transposed (d x kc); contiguous 16B store per lane per tile
    #pragma unroll
    for (int dt = 0; dt < 4; ++dt) {
      const int dcol = dt * 16 + ln;
      const int row0 = kl0 + ((lane & 16) ? 8 : 0);
      v8bf pk;
      #pragma unroll
      for (int r = 0; r < 8; ++r) pk[r] = (__bf16)acc[dt][r];
      *(uint4*)(Vct + ((size_t)bh * D_ + dcol) * KC_ + row0) =
          __builtin_bit_cast(uint4, pk);
    }
  }
}

// ---------------------------------------------------------------------------
// Gather: Kc rows 256..319 = K[sample_idxes], Vct cols 256..319 = V[sample_idxes].
// ---------------------------------------------------------------------------
__global__ __launch_bounds__(256) void gather_kernel(
    const float* __restrict__ K, const float* __restrict__ V,
    const int* __restrict__ sidx,
    unsigned short* __restrict__ Kc, unsigned short* __restrict__ Vct) {
  const int bh = blockIdx.x;
  for (int lin = threadIdx.x; lin < NS_ * D_; lin += 256) {
    const int j = lin >> 6;
    const int d = lin & 63;
    const int si = sidx[bh * NS_ + j];
    const size_t g = ((size_t)bh * S_ + si) * D_ + d;
    Kc[((size_t)bh * KC_ + KL_ + j) * D_ + d]   = f2bf(K[g]);
    Vct[((size_t)bh * D_ + d) * KC_ + KL_ + j]  = f2bf(V[g]);
  }
}

// ---------------------------------------------------------------------------
// Phase 2: scores = Q @ Kc^T / 8 ; softmax over 320 ; X = attn @ Vc + dwconv(Vm).
// Grid: B*H*(S/128) blocks, 256 threads = 8 waves, 16 queries per wave.
// ---------------------------------------------------------------------------
__global__ __launch_bounds__(256) void attn_kernel(
    const float* __restrict__ Q, const float* __restrict__ V,
    const float* __restrict__ mask, const float* __restrict__ cw,
    const unsigned short* __restrict__ Kc, const unsigned short* __restrict__ Vct,
    float* __restrict__ out) {
  __shared__ unsigned short ldsA[8][16 * 32];   // per-wave attn re-layout tile

  const int bh   = blockIdx.x >> 5;             // 32 q-blocks of 128
  const int qblk = blockIdx.x & 31;
  const int b = bh / H_;
  const int h = bh - b * H_;
  const int wave = threadIdx.x >> 5;
  const int lane = threadIdx.x & 31;
  const int q0 = qblk * 128 + wave * 16;
  const int ln = lane & 15;
  const int k0 = (lane >> 4) * 8;
  const int hi = (lane & 16) ? 1 : 0;

  const float* __restrict__ Qbh = Q + (size_t)bh * S_ * D_;
  const unsigned short* __restrict__ KcBH  = Kc  + (size_t)bh * KC_ * D_;
  const unsigned short* __restrict__ VctBH = Vct + (size_t)bh * D_ * KC_;

  // Q fragments for the two 32-wide d chunks
  v16bf aq0 = ld_afrag_f32(Qbh + (size_t)(q0 + ln) * D_ + 0  + k0);
  v16bf aq1 = ld_afrag_f32(Qbh + (size_t)(q0 + ln) * D_ + 32 + k0);

  // scores: 20 tiles of 16 kc columns, fp32 accumulate
  v8f sc[20];
  #pragma unroll
  for (int t = 0; t < 20; ++t) {
    const unsigned short* p = KcBH + (size_t)(t * 16 + ln) * D_ + (hi ? 8 : 0);
    v8f a = {};
    a = wmma_bf16(aq0, ld_bfrag_bf16(p),      a);
    a = wmma_bf16(aq1, ld_bfrag_bf16(p + 32), a);
    sc[t] = a;
  }

  // softmax over 320 per query row (rows live in 16-lane halves)
  float inv[8];
  #pragma unroll
  for (int r = 0; r < 8; ++r) {
    float mx = -3.0e38f;
    #pragma unroll
    for (int t = 0; t < 20; ++t) mx = fmaxf(mx, sc[t][r]);
    #pragma unroll
    for (int off = 1; off < 16; off <<= 1)
      mx = fmaxf(mx, __shfl_xor(mx, off, 32));
    float sm = 0.f;
    #pragma unroll
    for (int t = 0; t < 20; ++t) {
      const float e = __expf((sc[t][r] - mx) * 0.125f);   // 1/sqrt(64)
      sc[t][r] = e;
      sm += e;
    }
    #pragma unroll
    for (int off = 1; off < 16; off <<= 1)
      sm += __shfl_xor(sm, off, 32);
    inv[r] = 1.0f / sm;
  }

  // X = attn @ Vc ; re-layout attn C->A through per-wave LDS tile
  v8f xa[4] = {};
  unsigned short* myA = &ldsA[wave][0];
  #pragma unroll
  for (int c = 0; c < 10; ++c) {
    // B-fragments from Vct first (global; distinct regs, overlapped waits)
    v16bf bb[4];
    #pragma unroll
    for (int dt = 0; dt < 4; ++dt)
      bb[dt] = ld_bfrag_bf16(VctBH + (size_t)(dt * 16 + ln) * KC_ + 32 * c +
                             (hi ? 8 : 0));
    #pragma unroll
    for (int tt = 0; tt < 2; ++tt) {
      const int t = 2 * c + tt;
      #pragma unroll
      for (int r = 0; r < 8; ++r)
        myA[(r + hi * 8) * 32 + tt * 16 + ln] = f2bf(sc[t][r] * inv[r]);
    }
    // DS ops are in-order within a wave: safe to read back immediately
    const v16bf a = ld_afrag_bf16(myA + ln * 32 + k0);
    #pragma unroll
    for (int dt = 0; dt < 4; ++dt)
      xa[dt] = wmma_bf16(a, bb[dt], xa[dt]);
  }

  // epilogue: add depthwise conv of Vm (3-tap over s), store fp32
  const float w0 = cw[h * 3 + 0];
  const float w1 = cw[h * 3 + 1];
  const float w2 = cw[h * 3 + 2];
  #pragma unroll
  for (int dt = 0; dt < 4; ++dt) {
    const int d = dt * 16 + ln;
    #pragma unroll
    for (int r = 0; r < 8; ++r) {
      const int s = q0 + r + hi * 8;
      const size_t gi = ((size_t)bh * S_ + s) * D_ + d;
      float cv = w1 * V[gi] * mask[b * S_ + s];
      if (s > 0)       cv += w0 * V[gi - D_] * mask[b * S_ + s - 1];
      if (s < S_ - 1)  cv += w2 * V[gi + D_] * mask[b * S_ + s + 1];
      out[gi] = xa[dt][r] + cv;
    }
  }
}

// ---------------------------------------------------------------------------
extern "C" void kernel_launch(void* const* d_in, const int* in_sizes, int n_in,
                              void* d_out, int out_size, void* d_ws, size_t ws_size,
                              hipStream_t stream) {
  (void)in_sizes; (void)n_in; (void)out_size; (void)ws_size;
  const float* Q    = (const float*)d_in[0];
  const float* K    = (const float*)d_in[1];
  const float* V    = (const float*)d_in[2];
  const float* mask = (const float*)d_in[3];
  const float* E    = (const float*)d_in[4];
  const float* cw   = (const float*)d_in[5];
  const int*   sidx = (const int*)d_in[6];

  unsigned short* Kc  = (unsigned short*)d_ws;                       // B*H*320*64 bf16
  unsigned short* Vct = Kc + (size_t)B_ * H_ * KC_ * D_;             // B*H*64*320 bf16

  proj_kernel  <<<dim3(B_ * H_ * 4),          256, 0, stream>>>(K, V, mask, E, Kc, Vct);
  gather_kernel<<<dim3(B_ * H_),              256, 0, stream>>>(K, V, sidx, Kc, Vct);
  attn_kernel  <<<dim3(B_ * H_ * (S_ / 128)), 256, 0, stream>>>(Q, V, mask, cw, Kc, Vct,
                                                                (float*)d_out);
}
